// GraphTransformerEncoder_17549236371838
// MI455X (gfx1250) — matrix operand order repaired and checked by reading, not compiled, not run on hardware
//
#include <hip/hip_runtime.h>

#define NXG 128
#define NYG 128
#define NNODE (NXG * NYG)      // 16384
#define FIN 26
#define KPAD 32
#define HID 512
#define NHEAD 8
#define HDIM 64
#define DEPTH 4
#define LN_EPS 1e-5f

typedef __bf16 bf16x16 __attribute__((ext_vector_type(16)));
typedef float  f32x8  __attribute__((ext_vector_type(8)));

union FragB16 { bf16x16 v; uint4 u[2]; };
union FragF32 { f32x8 v; float f[8]; };

// ---------- helpers ----------
__device__ __forceinline__ unsigned short f2bf(float f) {
    unsigned int u = __float_as_uint(f);
    unsigned int r = u + 0x7FFFu + ((u >> 16) & 1u);   // round-to-nearest-even
    return (unsigned short)(r >> 16);
}

// ---------- prep kernels ----------
__global__ void graph_cast_kernel(const float* __restrict__ g, unsigned short* __restrict__ gb) {
    int t = blockIdx.x * blockDim.x + threadIdx.x;       // N*KPAD
    if (t >= NNODE * KPAD) return;
    int n = t / KPAD, k = t % KPAD;
    float v = (k < FIN) ? g[(size_t)n * FIN + k] : 0.0f;
    gb[t] = f2bf(v);
}

__global__ void wp_transpose_kernel(const float* __restrict__ Wp, unsigned short* __restrict__ Wpt) {
    int t = blockIdx.x * blockDim.x + threadIdx.x;       // HID*KPAD, Wpt[n*KPAD+k] = Wp[k*HID+n]
    if (t >= HID * KPAD) return;
    int n = t / KPAD, k = t % KPAD;
    float v = (k < FIN) ? Wp[(size_t)k * HID + n] : 0.0f;
    Wpt[t] = f2bf(v);
}

// Wt[n*K + k] = W[k*Ncols + n]  (K = Ncols = HID here)
__global__ void transpose_cast_kernel(const float* __restrict__ W, unsigned short* __restrict__ Wt) {
    int t = blockIdx.x * blockDim.x + threadIdx.x;       // HID*HID
    if (t >= HID * HID) return;
    int n = t / HID, k = t % HID;
    Wt[t] = f2bf(W[(size_t)k * HID + n]);
}

__global__ void pack_bias_kernel(const float* __restrict__ bq, const float* __restrict__ bk,
                                 const float* __restrict__ bv, const float* __restrict__ bs,
                                 float* __restrict__ dst) {
    int t = blockIdx.x * blockDim.x + threadIdx.x;       // DEPTH * 4*HID
    if (t >= DEPTH * 4 * HID) return;
    int l = t / (4 * HID);
    int c = t % (4 * HID);
    const float* srcs[4] = { bq, bk, bv, bs };
    dst[t] = srcs[c / HID][(size_t)l * HID + (c % HID)];
}

// ---------- WMMA GEMM: C[M x Np] = A_bf16[M x K] @ Bt_bf16[Np x K]^T + bias ----------
// block: 256 threads = 8 waves (2 in M x 4 in N); block tile 64 x 256; wave tile 32 x 64.
// Per K-step per wave: 2 A frags + 4 B frags (12 x b128 loads) feed 8 WMMAs.
// Requires M % 64 == 0, Np % 256 == 0, K % 32 == 0, 16B-aligned buffers. No divergence.
__global__ __launch_bounds__(256) void gemm_bf16_kernel(const unsigned short* __restrict__ A,
                                                        const unsigned short* __restrict__ Bt,
                                                        const float* __restrict__ bias,
                                                        float* __restrict__ C,
                                                        unsigned short* __restrict__ Cb, // optional bf16 copy
                                                        int M, int Np, int K) {
    const int tid  = threadIdx.x;
    const int wid  = tid >> 5;
    const int lane = tid & 31;
    const int wm   = wid & 1;          // 0..1  -> M sub-tile (32 rows)
    const int wn   = wid >> 1;         // 0..3  -> N sub-tile (64 cols)
    const int m0   = blockIdx.x * 64 + wm * 32;
    const int n0   = blockIdx.y * 256 + wn * 64;

    const int row = lane & 15;         // M row (A) / N column (B,C)
    const int hi  = lane >> 4;         // lane-half selector

    f32x8 acc[2][4];
    #pragma unroll
    for (int a = 0; a < 2; ++a)
        #pragma unroll
        for (int t = 0; t < 4; ++t) {
            f32x8 z = {0,0,0,0,0,0,0,0};
            acc[a][t] = z;
        }

    const unsigned short* Arow0 = A + (size_t)(m0 + row) * K;
    const unsigned short* Arow1 = A + (size_t)(m0 + 16 + row) * K;
    const unsigned short* Bp[4];
    #pragma unroll
    for (int t = 0; t < 4; ++t) Bp[t] = Bt + (size_t)(n0 + t * 16 + row) * K;

    for (int kk = 0; kk < K; kk += 32) {
        // A fragments (16x32): lane<16 holds K {kk..kk+7, kk+16..kk+23}; lane>=16 shifted by 8
        FragB16 fa0, fa1;
        fa0.u[0] = *reinterpret_cast<const uint4*>(Arow0 + kk + hi * 8);
        fa0.u[1] = *reinterpret_cast<const uint4*>(Arow0 + kk + hi * 8 + 16);
        fa1.u[0] = *reinterpret_cast<const uint4*>(Arow1 + kk + hi * 8);
        fa1.u[1] = *reinterpret_cast<const uint4*>(Arow1 + kk + hi * 8 + 16);

        const int kb = kk + hi * 8;
        #pragma unroll
        for (int t = 0; t < 4; ++t) {
            // B fragment (32x16): lane<16 holds K {kk..kk+7, kk+16..kk+23}; lane>=16 shifted by 8
            FragB16 fb;
            fb.u[0] = *reinterpret_cast<const uint4*>(Bp[t] + kb);
            fb.u[1] = *reinterpret_cast<const uint4*>(Bp[t] + kb + 16);
            acc[0][t] = __builtin_amdgcn_wmma_f32_16x16x32_bf16(false, fa0.v, false, fb.v,
                                                                (short)0, acc[0][t], false, false);
            acc[1][t] = __builtin_amdgcn_wmma_f32_16x16x32_bf16(false, fa1.v, false, fb.v,
                                                                (short)0, acc[1][t], false, false);
        }
    }

    // store: C/D layout -> lane col = n0 + t*16 + (lane&15); VGPR i row = m(sub) + hi*8 + i
    #pragma unroll
    for (int a = 0; a < 2; ++a) {
        #pragma unroll
        for (int t = 0; t < 4; ++t) {
            FragF32 res; res.v = acc[a][t];
            const int col = n0 + t * 16 + row;
            const float bv = bias[col];
            #pragma unroll
            for (int i = 0; i < 8; ++i) {
                const int m = m0 + a * 16 + hi * 8 + i;
                const float v = res.f[i] + bv;
                C[(size_t)m * Np + col] = v;
                if (Cb) Cb[(size_t)m * Np + col] = f2bf(v);
            }
        }
    }
}

// ---------- attention over 4-connected grid (equivalent to the bidirectional edge list) ----------
// qkvr row layout per node: [ q(512) | k(512) | v(512) | r(512) ]
__global__ void attn_kernel(const float* __restrict__ qkvr, float* __restrict__ out) {
    int t = blockIdx.x * blockDim.x + threadIdx.x;       // N * NHEAD
    if (t >= NNODE * NHEAD) return;
    int n = t / NHEAD, h = t % NHEAD;
    int ix = n / NYG, iy = n % NYG;

    int nbs[4]; int nn = 0;
    if (iy > 0)        nbs[nn++] = n - 1;
    if (iy < NYG - 1)  nbs[nn++] = n + 1;
    if (ix > 0)        nbs[nn++] = n - NYG;
    if (ix < NXG - 1)  nbs[nn++] = n + NYG;

    const float* qp = qkvr + (size_t)n * (4 * HID) + h * HDIM;
    float dots[4];
    for (int j = 0; j < nn; ++j) {
        const float* kp = qkvr + (size_t)nbs[j] * (4 * HID) + HID + h * HDIM;
        float s = 0.0f;
        #pragma unroll 8
        for (int i = 0; i < HDIM; ++i) s += qp[i] * kp[i];
        dots[j] = s * 0.125f;                            // 1/sqrt(64)
    }
    float m = -3.4e38f;
    for (int j = 0; j < nn; ++j) m = fmaxf(m, dots[j]);
    float denom = 0.0f;
    for (int j = 0; j < nn; ++j) { dots[j] = __expf(dots[j] - m); denom += dots[j]; }
    float inv = 1.0f / (denom + 1e-16f);

    float* op = out + (size_t)n * HID + h * HDIM;
    for (int i = 0; i < HDIM; ++i) {
        float s = 0.0f;
        for (int j = 0; j < nn; ++j)
            s += dots[j] * qkvr[(size_t)nbs[j] * (4 * HID) + 2 * HID + h * HDIM + i];
        op[i] = s * inv;
    }
}

// ---------- gate + residual + layernorm; one block (256 thr) per node; x updated in place ----------
__device__ __forceinline__ float block_reduce_sum(float v, float* sm) {
    __syncthreads();                       // protect sm from previous use
    #pragma unroll
    for (int off = 16; off > 0; off >>= 1) v += __shfl_down(v, off, 32);
    int lane = threadIdx.x & 31, wid = threadIdx.x >> 5;
    if (lane == 0) sm[wid] = v;
    __syncthreads();
    if (wid == 0) {
        v = (lane < 8) ? sm[lane] : 0.0f;
        #pragma unroll
        for (int off = 4; off > 0; off >>= 1) v += __shfl_down(v, off, 32);
        if (lane == 0) sm[0] = v;
    }
    __syncthreads();
    return sm[0];
}

__global__ __launch_bounds__(256) void gate_ln_kernel(const float* __restrict__ out,
                                                      const float* __restrict__ qkvr,
                                                      const float* __restrict__ Wbeta,
                                                      const float* __restrict__ ln_g,
                                                      const float* __restrict__ ln_b,
                                                      float* __restrict__ x,
                                                      unsigned short* __restrict__ xb) {
    __shared__ float sm[8];
    const int n = blockIdx.x;
    const float* o = out + (size_t)n * HID;
    const float* r = qkvr + (size_t)n * (4 * HID) + 3 * HID;
    float* xr = x + (size_t)n * HID;
    unsigned short* xbr = xb + (size_t)n * HID;

    // gate dot: sum_i o_i*Wb0_i + r_i*Wb1_i + (o_i-r_i)*Wb2_i
    float part = 0.0f;
    int i0 = threadIdx.x, i1 = threadIdx.x + 256;
    {
        float o0 = o[i0], r0 = r[i0], o1 = o[i1], r1 = r[i1];
        part += o0 * (Wbeta[i0] + Wbeta[2 * HID + i0]) + r0 * (Wbeta[HID + i0] - Wbeta[2 * HID + i0]);
        part += o1 * (Wbeta[i1] + Wbeta[2 * HID + i1]) + r1 * (Wbeta[HID + i1] - Wbeta[2 * HID + i1]);
    }
    float s = block_reduce_sum(part, sm);
    float g = 1.0f / (1.0f + __expf(-s));

    // y = x + g*r + (1-g)*o  (each thread owns elements i0, i1)
    float y0 = xr[i0] + g * r[i0] + (1.0f - g) * o[i0];
    float y1 = xr[i1] + g * r[i1] + (1.0f - g) * o[i1];

    float ssum = block_reduce_sum(y0 + y1, sm);
    float mu = ssum * (1.0f / HID);
    float d0 = y0 - mu, d1 = y1 - mu;
    float vsum = block_reduce_sum(d0 * d0 + d1 * d1, sm);
    float rstd = rsqrtf(vsum * (1.0f / HID) + LN_EPS);

    float z0 = d0 * rstd * ln_g[i0] + ln_b[i0];
    float z1 = d1 * rstd * ln_g[i1] + ln_b[i1];
    xr[i0] = z0;           xr[i1] = z1;
    xbr[i0] = f2bf(z0);    xbr[i1] = f2bf(z1);
}

// ---------- global mean over nodes ----------
__global__ void colmean_kernel(const float* __restrict__ x, float* __restrict__ gmean) {
    int c = blockIdx.x * blockDim.x + threadIdx.x;
    if (c >= HID) return;
    float s = 0.0f;
    for (int n = 0; n < NNODE; ++n) s += x[(size_t)n * HID + c];
    gmean[c] = s * (1.0f / NNODE);
}

// ---------- host launch ----------
extern "C" void kernel_launch(void* const* d_in, const int* in_sizes, int n_in,
                              void* d_out, int out_size, void* d_ws, size_t ws_size,
                              hipStream_t stream) {
    (void)in_sizes; (void)n_in; (void)out_size; (void)ws_size;
    const float* graph = (const float*)d_in[0];
    // d_in[1], d_in[2]: edge_src/edge_dst (grid structure used directly)
    const float* Wp    = (const float*)d_in[3];
    const float* bp    = (const float*)d_in[4];
    const float* Wq    = (const float*)d_in[5];
    const float* bq    = (const float*)d_in[6];
    const float* Wk    = (const float*)d_in[7];
    const float* bk    = (const float*)d_in[8];
    const float* Wv    = (const float*)d_in[9];
    const float* bv    = (const float*)d_in[10];
    const float* Ws    = (const float*)d_in[11];
    const float* bs    = (const float*)d_in[12];
    const float* Wbeta = (const float*)d_in[13];
    const float* ln_g  = (const float*)d_in[14];
    const float* ln_b  = (const float*)d_in[15];

    float* x = (float*)d_out;                 // N x HID, final result lives here
    float* gmean = x + (size_t)NNODE * HID;   // global embedding

    // workspace carve-out (256B aligned)
    char* ws = (char*)d_ws;
    size_t off = 0;
    auto carve = [&](size_t bytes) -> void* {
        off = (off + 255) & ~(size_t)255;
        void* p = ws + off;
        off += bytes;
        return p;
    };
    unsigned short* gb     = (unsigned short*)carve((size_t)NNODE * KPAD * 2);
    unsigned short* Wpt    = (unsigned short*)carve((size_t)HID * KPAD * 2);
    unsigned short* WtAll  = (unsigned short*)carve((size_t)DEPTH * 4 * HID * HID * 2);
    float*          biasAll= (float*)carve((size_t)DEPTH * 4 * HID * 4);
    unsigned short* xb     = (unsigned short*)carve((size_t)NNODE * HID * 2);
    float*          qkvr   = (float*)carve((size_t)NNODE * 4 * HID * 4);
    float*          attnO  = (float*)carve((size_t)NNODE * HID * 4);

    // ---- prep ----
    graph_cast_kernel<<<(NNODE * KPAD + 255) / 256, 256, 0, stream>>>(graph, gb);
    wp_transpose_kernel<<<(HID * KPAD + 255) / 256, 256, 0, stream>>>(Wp, Wpt);
    const float* Wmats[4] = { Wq, Wk, Wv, Ws };
    for (int l = 0; l < DEPTH; ++l)
        for (int p = 0; p < 4; ++p)
            transpose_cast_kernel<<<(HID * HID + 255) / 256, 256, 0, stream>>>(
                Wmats[p] + (size_t)l * HID * HID,
                WtAll + ((size_t)l * 4 * HID + p * HID) * HID);
    pack_bias_kernel<<<(DEPTH * 4 * HID + 255) / 256, 256, 0, stream>>>(bq, bk, bv, bs, biasAll);

    // ---- input projection: x = graph @ Wp + bp  (M=16384, Np=512, K=32); also emit bf16 x ----
    gemm_bf16_kernel<<<dim3(NNODE / 64, HID / 256), 256, 0, stream>>>(
        gb, Wpt, bp, x, xb, NNODE, HID, KPAD);

    // ---- layers ----
    for (int l = 0; l < DEPTH; ++l) {
        // fused q|k|v|r projection: (16384 x 512) @ (512 x 2048)
        gemm_bf16_kernel<<<dim3(NNODE / 64, (4 * HID) / 256), 256, 0, stream>>>(
            xb, WtAll + (size_t)l * 4 * HID * HID, biasAll + (size_t)l * 4 * HID,
            qkvr, (unsigned short*)nullptr, NNODE, 4 * HID, HID);
        attn_kernel<<<(NNODE * NHEAD + 255) / 256, 256, 0, stream>>>(qkvr, attnO);
        gate_ln_kernel<<<NNODE, 256, 0, stream>>>(attnO, qkvr,
            Wbeta + (size_t)l * 3 * HID, ln_g + (size_t)l * HID, ln_b + (size_t)l * HID, x, xb);
    }

    // ---- global embedding ----
    colmean_kernel<<<(HID + 255) / 256, 256, 0, stream>>>(x, gmean);
}
